// DecoderRNN_23922967839057
// MI455X (gfx1250) — compile-verified, matrix operand order checked by reading, not run on hardware
//
#include <hip/hip_runtime.h>
#include <hip/hip_bf16.h>
#include <stdint.h>

// Problem dims (fixed by reference)
#define B_   32
#define T_   64
#define E_   512
#define H_   512
#define K_   512            // shared K for both big GEMMs (E_ == H_)
#define V_   32000
#define G4H  (4 * H_)       // 2048
#define TB   (T_ * B_)      // 2048

typedef __attribute__((ext_vector_type(16))) _Float16 v16h;
typedef __attribute__((ext_vector_type(8)))  _Float16 v8h;
typedef __attribute__((ext_vector_type(8)))  float    v8f;
typedef __attribute__((ext_vector_type(4)))  int      v4i;

// ---------------------------------------------------------------------------
// gfx1250 async global->LDS path (ASYNCcnt-tracked). The builtin is typed
// (v4i as1*, v4i as3*, imm offset, imm cpol) per the compiler diagnostic.
// Guarded so the file still compiles if the builtin is absent (falls back to
// load + ds_store; same LDS reuse, no async overlap).
// ---------------------------------------------------------------------------
#if defined(__has_builtin)
#  if __has_builtin(__builtin_amdgcn_global_load_async_to_lds_b128)
#    define USE_ASYNC_LDS 1
#  endif
#endif
#ifndef USE_ASYNC_LDS
#  define USE_ASYNC_LDS 0
#endif

#if USE_ASYNC_LDS
typedef __attribute__((address_space(1))) v4i* gptr_v4i;
typedef __attribute__((address_space(3))) v4i* lptr_v4i;

__device__ __forceinline__ void async_copy_b128(void* lds, const void* g) {
  // generic->AS1 keeps the 64-bit address; generic->AS3 is the low-32-bit
  // LDS offset (matches LLVM's addrspacecast semantics), done via integers
  // so no pointer-addrspace conversion rules get in the way.
  __builtin_amdgcn_global_load_async_to_lds_b128(
      (gptr_v4i)(uintptr_t)g,
      (lptr_v4i)(uint32_t)(uintptr_t)lds,
      0, 0);
}
__device__ __forceinline__ void wait_async_le1() {
#  if __has_builtin(__builtin_amdgcn_s_wait_asynccnt)
  __builtin_amdgcn_s_wait_asynccnt(1);
#  else
  asm volatile("s_wait_asynccnt 0x1" ::: "memory");
#  endif
}
__device__ __forceinline__ void wait_async_le0() {
#  if __has_builtin(__builtin_amdgcn_s_wait_asynccnt)
  __builtin_amdgcn_s_wait_asynccnt(0);
#  else
  asm volatile("s_wait_asynccnt 0x0" ::: "memory");
#  endif
}
#endif

// ---------------------------------------------------------------------------
// WMMA fragment builder. CDNA5 16-bit 16x32 A layout (ISA 7.12.2):
// lane L -> row m = L%16, elements [0..7] = K[8*(L/16)..+7],
// elements [8..15] = K[16+8*(L/16)..+7]. B (KxN, stored row-major [N][K])
// uses the identical per-lane packing with the lane selecting column N.
// Every fragment is two contiguous 16B loads (global_load_b128 or
// ds_load_b128 depending on pointer address space).
// ---------------------------------------------------------------------------
__device__ __forceinline__ v16h make_frag(const _Float16* p) {
  v8h lo = *(const v8h*)p;
  v8h hi = *(const v8h*)(p + 16);
  return __builtin_shufflevector(lo, hi, 0, 1, 2, 3, 4, 5, 6, 7,
                                         8, 9, 10, 11, 12, 13, 14, 15);
}

__device__ __forceinline__ v8f wmma_f16(v16h a, v16h b, v8f c) {
  return __builtin_amdgcn_wmma_f32_16x16x32_f16(false, a, false, b,
                                                (short)0, c, false, false);
}

// ---------------------------------------------------------------------------
// Prep kernels
// ---------------------------------------------------------------------------
__global__ void k_f32_to_f16(const float* __restrict__ src,
                             _Float16* __restrict__ dst, int n) {
  int i = blockIdx.x * blockDim.x + threadIdx.x;
  int stride = gridDim.x * blockDim.x;
  for (; i < n; i += stride) dst[i] = (_Float16)src[i];
}

// X[t*B+b][e] : t==0 -> features[b][e]; t>0 -> emb[captions[b][t]][e]
__global__ void k_build_x(const float* __restrict__ features,
                          const int* __restrict__ captions,
                          const float* __restrict__ emb,
                          _Float16* __restrict__ X) {
  int idx = blockIdx.x * blockDim.x + threadIdx.x;
  if (idx >= TB * E_) return;
  int e  = idx & (E_ - 1);
  int tb = idx >> 9;          // E_ == 512
  int b  = tb & (B_ - 1);
  int t  = tb >> 5;           // B_ == 32
  float v;
  if (t == 0) {
    v = features[b * E_ + e];
  } else {
    int cap = captions[b * T_ + t];
    v = emb[(size_t)cap * E_ + e];
  }
  X[idx] = (_Float16)v;
}

// ---------------------------------------------------------------------------
// Blocked WMMA GEMM:  C[2048 x N] = A[2048 x 512] @ B[N x 512]^T + bias
// Workgroup (8 wave32) computes 64 rows x 256 cols:
//   - 64x32 A K-slab double-buffered in LDS (async global->LDS, one b128
//     per thread), shared by all 8 waves -> ds_load_b128 fragment reads.
//   - each wave owns 2 N-tiles (32 cols); B fragments stream from global.
//   - register blocking: v8f acc[4][2] = 8 WMMAs per 32-K chunk per wave.
// Epilogue: PERMUTE=false -> C[m*ldc + n] (gates_x);
//           PERMUTE=true  -> logits[b][t][v] with m = t*B+b (output GEMM).
// ---------------------------------------------------------------------------
#define LDA_LDS 40   // 32-halves K-slab row + 8-halves pad (80B, 16B-aligned)

template <bool PERMUTE>
__global__ void __launch_bounds__(256)
k_gemm_blocked(const _Float16* __restrict__ A,    // [2048, 512] row-major
               const _Float16* __restrict__ Bm,   // [N, 512] row-major
               const float* __restrict__ bias1,
               const float* __restrict__ bias2,   // nullable
               float* __restrict__ C, int ldc) {
  __shared__ _Float16 sA[2][64 * LDA_LDS];        // 2 x 5 KB

  const int lane = threadIdx.x & 31;
  const int wave = threadIdx.x >> 5;
  const int half = lane >> 4;
  const int l16  = lane & 15;
  const int mblock = blockIdx.y * 64;
  const int nblock = blockIdx.x * 256;

  // A staging: 64 rows x 32 halves = 256 x 16B segments, one per thread.
  const int srow = threadIdx.x >> 2;              // 0..63
  const int sseg = (threadIdx.x & 3) * 8;         // halves: 0,8,16,24
  const _Float16* gA = A + (size_t)(mblock + srow) * K_ + sseg;
  _Float16* lA0 = &sA[0][srow * LDA_LDS + sseg];
  _Float16* lA1 = &sA[1][srow * LDA_LDS + sseg];

  // Wave's two B column-tiles.
  const _Float16* bp0 = Bm + (size_t)(nblock + wave * 32 + l16) * K_ + 8 * half;
  const _Float16* bp1 = bp0 + (size_t)16 * K_;

  v8f acc[4][2] = {};

  // Prologue: stage first K-slab into buffer 0.
#if USE_ASYNC_LDS
  async_copy_b128(lA0, gA);
#else
  *(v8h*)lA0 = *(const v8h*)gA;
#endif

  int buf = 0;
  for (int kb = 0; kb < K_; kb += 32) {
    const bool has_next = (kb + 32) < K_;
    _Float16* lnext = (buf ? lA0 : lA1);
    if (has_next) {
#if USE_ASYNC_LDS
      async_copy_b128(lnext, gA + kb + 32);
#else
      *(v8h*)lnext = *(const v8h*)(gA + kb + 32);
#endif
    }
#if USE_ASYNC_LDS
    if (has_next) wait_async_le1();   // current slab landed, next in flight
    else          wait_async_le0();
#endif
    __syncthreads();

    v16h bf0 = make_frag(bp0 + kb);
    v16h bf1 = make_frag(bp1 + kb);
#pragma unroll
    for (int mt = 0; mt < 4; ++mt) {
      const _Float16* ap = &sA[buf][(mt * 16 + l16) * LDA_LDS + 8 * half];
      v16h af = make_frag(ap);        // 2 x ds_load_b128
      acc[mt][0] = wmma_f16(af, bf0, acc[mt][0]);
      acc[mt][1] = wmma_f16(af, bf1, acc[mt][1]);
    }
    __syncthreads();                  // reads done before slab is overwritten
    buf ^= 1;
  }

  // Epilogue
#pragma unroll
  for (int mt = 0; mt < 4; ++mt) {
#pragma unroll
    for (int nt = 0; nt < 2; ++nt) {
      const int n = nblock + wave * 32 + nt * 16 + l16;
      const float bs = bias1[n] + (bias2 ? bias2[n] : 0.0f);
#pragma unroll
      for (int r = 0; r < 8; ++r) {
        const int m = mblock + mt * 16 + r + 8 * half;
        const float v = acc[mt][nt][r] + bs;
        if (PERMUTE) {
          const int t = m >> 5;               // m = t*B + b
          const int b = m & (B_ - 1);
          C[((size_t)b * T_ + t) * V_ + n] = v;
        } else {
          C[(size_t)m * ldc + n] = v;
        }
      }
    }
  }
}

// ---------------------------------------------------------------------------
// Per-step recurrent GEMM: gates = gates_x[t] + Hprev @ W_hh^T   [B, 4H]
// Tiny and latency-bound: 256 tiles, one per wave, 32 blocks of 8 waves.
// ---------------------------------------------------------------------------
__global__ void k_gemm_step(const _Float16* __restrict__ Hprev,   // [B, H]
                            const _Float16* __restrict__ Whh,     // [4H, H]
                            const float* __restrict__ gates_x_t,  // [B, 4H]
                            float* __restrict__ gates) {          // [B, 4H]
  const int lane = threadIdx.x & 31;
  const int wave = threadIdx.x >> 5;
  const int tile = blockIdx.x * 8 + wave;   // 0..255
  const int mbase = (tile >> 7) * 16;       // 0 or 16
  const int nbase = (tile & 127) * 16;
  const int half = lane >> 4;
  const int l16  = lane & 15;
  const _Float16* ap = Hprev + (size_t)(mbase + l16) * H_ + 8 * half;
  const _Float16* bp = Whh   + (size_t)(nbase + l16) * H_ + 8 * half;
  v8f acc = {};
  for (int kb = 0; kb < H_; kb += 32) {
    acc = wmma_f16(make_frag(ap + kb), make_frag(bp + kb), acc);
  }
  const int n = nbase + l16;
#pragma unroll
  for (int r = 0; r < 8; ++r) {
    int m = mbase + r + 8 * half;
    size_t o = (size_t)m * G4H + n;
    gates[o] = gates_x_t[o] + acc[r];
  }
}

// ---------------------------------------------------------------------------
// Pointwise LSTM cell update (gate order i,f,g,o). Writes h as f16.
// ---------------------------------------------------------------------------
__global__ void k_lstm_pointwise(const float* __restrict__ gates, // [B, 4H]
                                 float* __restrict__ c,           // [B, H]
                                 _Float16* __restrict__ h_out,    // [B, H]
                                 int is_first) {
  int idx = blockIdx.x * blockDim.x + threadIdx.x;
  if (idx >= B_ * H_) return;
  int b = idx >> 9;             // H_ == 512
  int j = idx & (H_ - 1);
  const float* g = gates + (size_t)b * G4H;
  float i_ = 1.0f / (1.0f + __expf(-g[j]));
  float f_ = 1.0f / (1.0f + __expf(-g[H_ + j]));
  float g_ = tanhf(g[2 * H_ + j]);
  float o_ = 1.0f / (1.0f + __expf(-g[3 * H_ + j]));
  float cp = is_first ? 0.0f : c[idx];
  float cn = f_ * cp + i_ * g_;
  c[idx] = cn;
  h_out[idx] = (_Float16)(o_ * tanhf(cn));
}

// ---------------------------------------------------------------------------
extern "C" void kernel_launch(void* const* d_in, const int* in_sizes, int n_in,
                              void* d_out, int out_size, void* d_ws,
                              size_t ws_size, hipStream_t stream) {
  (void)in_sizes; (void)n_in; (void)out_size; (void)ws_size;
  const float* features = (const float*)d_in[0];   // [B, E]
  const int*   captions = (const int*)d_in[1];     // [B, T]
  const float* W_ih     = (const float*)d_in[2];   // [4H, E]
  const float* W_hh     = (const float*)d_in[3];   // [4H, H]
  const float* b_ih     = (const float*)d_in[4];
  const float* b_hh     = (const float*)d_in[5];
  const float* W_out    = (const float*)d_in[6];   // [V, H]
  const float* b_out    = (const float*)d_in[7];
  const float* emb      = (const float*)d_in[8];   // [V, E]
  float* out = (float*)d_out;                      // [B, T, V]

  // Workspace carve-up (256B aligned)
  char* ws = (char*)d_ws;
  size_t off = 0;
  auto carve = [&](size_t bytes) -> void* {
    void* p = ws + off;
    off += (bytes + 255) & ~(size_t)255;
    return p;
  };
  _Float16* Xf16    = (_Float16*)carve((size_t)TB * E_ * 2);     //  2 MB
  _Float16* Wih16   = (_Float16*)carve((size_t)G4H * E_ * 2);    //  2 MB
  _Float16* Whh16   = (_Float16*)carve((size_t)G4H * H_ * 2);    //  2 MB
  _Float16* Wout16  = (_Float16*)carve((size_t)V_ * H_ * 2);     // 32 MB
  float*    gates_x = (float*)carve((size_t)TB * G4H * 4);       // 16 MB
  _Float16* hs16    = (_Float16*)carve((size_t)TB * H_ * 2);     //  2 MB
  float*    cstate  = (float*)carve((size_t)B_ * H_ * 4);
  float*    gbuf    = (float*)carve((size_t)B_ * G4H * 4);

  // 1) Convert weights to f16 + build the input sequence (gather).
  k_f32_to_f16<<<2048, 256, 0, stream>>>(W_ih, Wih16, G4H * E_);
  k_f32_to_f16<<<2048, 256, 0, stream>>>(W_hh, Whh16, G4H * H_);
  k_f32_to_f16<<<4096, 256, 0, stream>>>(W_out, Wout16, V_ * H_);
  k_build_x<<<(TB * E_) / 256, 256, 0, stream>>>(features, captions, emb, Xf16);

  // 2) All timesteps' input-gate contributions in one blocked WMMA GEMM.
  //    N = 4H = 2048 -> 8 N-blocks; M = 2048 -> 32 M-blocks.
  k_gemm_blocked<false><<<dim3(8, 32), 256, 0, stream>>>(
      Xf16, Wih16, b_ih, b_hh, gates_x, G4H);

  // 3) Sequential recurrence: tiny WMMA GEMM + pointwise cell per step.
  for (int t = 0; t < T_; ++t) {
    const float* gsrc;
    if (t == 0) {
      gsrc = gates_x;                 // h0 == 0 -> gates are gates_x row block
    } else {
      k_gemm_step<<<32, 256, 0, stream>>>(hs16 + (size_t)(t - 1) * B_ * H_,
                                          Whh16,
                                          gates_x + (size_t)t * B_ * G4H,
                                          gbuf);
      gsrc = gbuf;
    }
    k_lstm_pointwise<<<(B_ * H_) / 256, 256, 0, stream>>>(
        gsrc, cstate, hs16 + (size_t)t * B_ * H_, t == 0 ? 1 : 0);
  }

  // 4) Output projection (dominant 67-GFLOP GEMM), fused permute + bias.
  //    N = 32000 -> 125 N-blocks; M = 2048 -> 32 M-blocks.
  k_gemm_blocked<true><<<dim3(125, 32), 256, 0, stream>>>(
      hs16, Wout16, b_out, nullptr, out, V_);
}
